// KP_Decoder_19490561590185
// MI455X (gfx1250) — matrix operand order repaired and checked by reading, not compiled, not run on hardware
//
#include <hip/hip_runtime.h>
#include <math.h>
#include <stdint.h>

// ---------------------------------------------------------------- constants
constexpr int Bb   = 32;    // batch
constexpr int Gg   = 512;   // POMO rollouts
constexpr int Nn   = 512;   // nodes
constexpr int EMB  = 128;
constexpr int Hh   = 8;
constexpr int DK   = 16;
constexpr int Ee   = 8;     // experts
constexpr int FH   = 512;   // ff hidden
constexpr float INV_SQRT_EMB = 1.0f / 11.313708498984761f;
constexpr float CLIP = 10.0f;

typedef __attribute__((ext_vector_type(16))) _Float16 v16h;
typedef __attribute__((ext_vector_type(8)))  float    v8f;

#define WMMA16(a, b, c) \
  __builtin_amdgcn_wmma_f32_16x16x32_f16(false, (a), false, (b), (short)0, (c), false, false)

// ---- CDNA5 async global->LDS copy (ASYNCcnt path, bypasses VGPRs) ----
// the LDS destination address is taken from the VDST VGPR; generic LDS pointers carry the wave-relative
// offset in their low 32 bits (flat->LDS truncation, ISA 10.2).
static __device__ __forceinline__ void async_ld_b128(unsigned lds_off, const void* gaddr) {
  asm volatile("global_load_async_to_lds_b128 %0, %1, off"
               :: "v"(lds_off), "v"(gaddr) : "memory");
}
static __device__ __forceinline__ void wait_async0() {
  asm volatile("s_wait_asynccnt 0" ::: "memory");
}

// ---- fragment loaders (CDNA5 ISA 7.12.2 wave32 16-bit layouts) ----
// A 16x32 (MxK): lanes 0-15 rows M, VGPR j -> K = 8*kh + 2j (pairs), VGPR 4+j -> K = 16 + 8*kh + 2j
static __device__ __forceinline__ v16h ld_a16(const _Float16* A, int ldm, int kb, int lane) {
  int m = lane & 15, kh = lane >> 4;
  union { v16h v; unsigned u[8]; } r;
  const unsigned* row = (const unsigned*)(A + (size_t)m * ldm + kb);
#pragma unroll
  for (int j = 0; j < 4; ++j) { r.u[j] = row[kh * 4 + j]; r.u[4 + j] = row[8 + kh * 4 + j]; }
  return r.v;
}
// B 32x16 (KxN) sourced from a row-major B^T (i.e. src[n][k]): lane col n = lane&15,
// VGPR j -> K = 8*kh + 2j, VGPR 4+j -> K = 16 + 8*kh + 2j (same as A).
static __device__ __forceinline__ v16h ld_bt16(const _Float16* BT, int ldn, int kb, int lane) {
  int n = lane & 15, kh = lane >> 4;
  union { v16h v; unsigned u[8]; } r;
  const unsigned* row = (const unsigned*)(BT + (size_t)n * ldn + kb);
#pragma unroll
  for (int j = 0; j < 4; ++j) { r.u[j] = row[kh * 4 + j]; r.u[4 + j] = row[8 + kh * 4 + j]; }
  return r.v;
}

// ---------------------------------------------------------------- kernel 1: hypernet
__global__ void hyper_kernel(const float* pref, const float* fc1_w, const float* fc1_b,
                             const float* fc2_w, const float* fc2_b, const float* fc3_w,
                             const float* fc3_b, const float* hWq, const float* hWk,
                             const float* hWv, const float* hWc, const float* gate_pref_w,
                             float* mid_out, float* prefgate, float* Wq,
                             _Float16* Wk16, _Float16* Wv16, _Float16* Wc16) {
  __shared__ float h1[256], h2[256], mid[8];
  int tid = threadIdx.x;  // 256 threads
  float p0 = pref[0], p1 = pref[1];
  h1[tid] = fc1_w[tid * 2 + 0] * p0 + fc1_w[tid * 2 + 1] * p1 + fc1_b[tid];
  __syncthreads();
  float acc = fc2_b[tid];
  for (int j = 0; j < 256; ++j) acc += fc2_w[tid * 256 + j] * h1[j];
  h2[tid] = acc;
  __syncthreads();
  if (tid < 8) {
    float a = fc3_b[tid];
    for (int j = 0; j < 256; ++j) a += fc3_w[tid * 256 + j] * h2[j];
    mid[tid] = a;
    mid_out[tid] = a;
  }
  __syncthreads();
  float m0 = mid[0], m1 = mid[1], m2 = mid[2], m3 = mid[3];
  float m4 = mid[4], m5 = mid[5], m6 = mid[6], m7 = mid[7];
  for (int i = tid; i < (1 + EMB) * EMB; i += 256)
    Wq[i] = hWq[i * 2] * m0 + hWq[i * 2 + 1] * m1;
  for (int i = tid; i < EMB * EMB; i += 256) {
    Wk16[i] = (_Float16)(hWk[i * 2] * m2 + hWk[i * 2 + 1] * m3);
    Wv16[i] = (_Float16)(hWv[i * 2] * m4 + hWv[i * 2 + 1] * m5);
    Wc16[i] = (_Float16)(hWc[i * 2] * m6 + hWc[i * 2 + 1] * m7);
  }
  if (tid < Ee) {
    float a = 0.f;
    for (int i = 0; i < 8; ++i) a += mid[i] * gate_pref_w[i * Ee + tid];
    prefgate[tid] = a;
  }
}

// ---------------------------------------------------------------- kernel 2: qbase
__global__ void qbase_kernel(const float* Wq, const float* graph, float* qbase) {
  int b = blockIdx.x, d = threadIdx.x;  // <<<32,128>>>
  const float* gr = graph + (size_t)b * EMB;
  const float* wr = Wq + (size_t)d * (EMB + 1);
  float a = 0.f;
  for (int c = 0; c < EMB; ++c) a += wr[c] * gr[c];
  qbase[b * EMB + d] = a;
}

// ---------------------------------------------------------------- kernel 3: f32->f16 conversions + q
constexpr size_t ENC_CNT = (size_t)Bb * Nn * EMB;          // 2097152
constexpr size_t W1_CNT  = (size_t)Ee * FH * EMB;          // 524288
constexpr size_t W2_CNT  = (size_t)Ee * EMB * FH;          // 524288
constexpr size_t Q_CNT   = (size_t)Bb * Gg * EMB;          // 2097152
constexpr size_t CVT_TOTAL = ENC_CNT + W1_CNT + W2_CNT + Q_CNT;

__global__ void convert_kernel(const float* enc, const float* e_w1, const float* e_w2,
                               const float* qbase, const float* capacity, const float* Wq,
                               _Float16* enc16, _Float16* w1_16, _Float16* w2_16, _Float16* q16) {
  for (size_t i = (size_t)blockIdx.x * blockDim.x + threadIdx.x; i < CVT_TOTAL;
       i += (size_t)gridDim.x * blockDim.x) {
    if (i < ENC_CNT) {
      enc16[i] = (_Float16)enc[i];
    } else if (i < ENC_CNT + W1_CNT) {
      size_t j = i - ENC_CNT; w1_16[j] = (_Float16)e_w1[j];
    } else if (i < ENC_CNT + W1_CNT + W2_CNT) {
      size_t j = i - ENC_CNT - W1_CNT; w2_16[j] = (_Float16)e_w2[j];
    } else {
      size_t j = i - ENC_CNT - W1_CNT - W2_CNT;   // token*128 + d
      size_t t = j >> 7; int d = (int)(j & 127); int b = (int)(t >> 9);
      q16[j] = (_Float16)(qbase[b * EMB + d] + capacity[t] * Wq[d * (EMB + 1) + EMB]);
    }
  }
}

// ---------------------------------------------------------------- kernel 4: K/V projections (WMMA)
// K = enc16 @ Wk^T, V = enc16 @ Wv^T    rows = B*N = 16384, cols = 128, K-depth 128
__global__ void kv_kernel(const _Float16* enc16, const _Float16* Wk16, const _Float16* Wv16,
                          _Float16* K16, _Float16* V16) {
  int wid = blockIdx.x * (blockDim.x >> 5) + (threadIdx.x >> 5);
  int lane = threadIdx.x & 31;
  int mat  = wid >> 13;               // 8192 tiles per matrix
  int t    = wid & 8191;
  int rowt = t >> 3, colt = t & 7;
  const _Float16* W   = mat ? Wv16 : Wk16;
  _Float16*       out = mat ? V16 : K16;
  const _Float16* Arow = enc16 + (size_t)rowt * 16 * EMB;
  const _Float16* Brow = W + (size_t)colt * 16 * EMB;
  v8f c = {};
#pragma unroll
  for (int kb = 0; kb < EMB; kb += 32) {
    v16h a = ld_a16(Arow, EMB, kb, lane);
    v16h b = ld_bt16(Brow, EMB, kb, lane);
    c = WMMA16(a, b, c);
  }
  int m0 = (lane >> 4) * 8, n = lane & 15;
#pragma unroll
  for (int r = 0; r < 8; ++r)
    out[((size_t)(rowt * 16 + m0 + r)) * EMB + colt * 16 + n] = (_Float16)c[r];
}

// ---------------------------------------------------------------- kernel 5: fused attention
// one block per (b,h); 4 waves; LDS: k(16K) + vT(16K) + 4x score f32(128K) + 4x P f16(64K) = 224KB
__global__ void attn_kernel(const _Float16* q16, const _Float16* K16, const _Float16* V16,
                            const float* mask, _Float16* out16) {
  extern __shared__ char smem[];
  _Float16* ksh  = (_Float16*)smem;                    // [N][16]
  _Float16* vshT = ksh + Nn * DK;                      // [16][N]
  float*    sbuf = (float*)(vshT + DK * Nn);           // 4 * [16][N]
  _Float16* pbuf = (_Float16*)(sbuf + 4 * 16 * Nn);    // 4 * [16][N]
  int b = blockIdx.x >> 3, h = blockIdx.x & 7;
  int tid = threadIdx.x, wave = tid >> 5, lane = tid & 31;

  // K tile: contiguous 16B chunks -> async copy straight to LDS (ASYNCcnt)
  {
    unsigned ksh_off = (unsigned)(uintptr_t)ksh;
    const _Float16* Kbase = K16 + (size_t)b * Nn * EMB + h * DK;
    for (int c = tid; c < (Nn * DK * 2) / 16; c += blockDim.x) {  // 1024 chunks
      int n = c >> 1, half = c & 1;
      async_ld_b128(ksh_off + (unsigned)c * 16, Kbase + (size_t)n * EMB + half * 8);
    }
  }
  // V tile needs a 16-bit transpose -> manual path
  for (int i = tid; i < Nn * DK; i += blockDim.x) {
    int n = i >> 4, d = i & 15;
    vshT[d * Nn + n] = V16[((size_t)(b * Nn + n)) * EMB + h * DK + d];
  }
  wait_async0();
  __syncthreads();

  float*    ms = sbuf + wave * 16 * Nn;
  _Float16* mp = pbuf + wave * 16 * Nn;

  for (int gt = wave; gt < Gg / 16; gt += 4) {   // 8 uniform iterations per wave
    int g0 = gt * 16;
    // A = q tile 16x16 padded to K=32 (upper half zero)
    union { v16h v; unsigned u[8]; } aq;
    {
      int m = lane & 15, kh = lane >> 4;
      const unsigned* row =
          (const unsigned*)(q16 + ((size_t)(b * Gg + g0 + m)) * EMB + h * DK);
#pragma unroll
      for (int j = 0; j < 4; ++j) { aq.u[j] = row[kh * 4 + j]; aq.u[4 + j] = 0u; }
    }
    // score tiles
    for (int nt = 0; nt < Nn / 16; ++nt) {
      int n0 = nt * 16;
      union { v16h v; unsigned u[8]; } bk;
      int ncol = lane & 15, khB = lane >> 4;
      const unsigned* rowB = (const unsigned*)(ksh + (n0 + ncol) * DK);
#pragma unroll
      for (int j = 0; j < 8; ++j) bk.u[j] = (j < 4) ? rowB[khB * 4 + j] : 0u;
      v8f c = {};
      c = WMMA16(aq.v, bk.v, c);
      int mrow = (lane >> 4) * 8;
#pragma unroll
      for (int r = 0; r < 8; ++r) {
        int m2 = mrow + r;
        ms[m2 * Nn + n0 + ncol] =
            c[r] * 0.25f + mask[((size_t)(b * Gg + g0 + m2)) * Nn + n0 + ncol];
      }
    }
    __syncthreads();
    // softmax: 2 lanes per row (halves of 256 columns each)
    {
      int row = lane & 15, half = lane >> 4;
      float* rp = ms + row * Nn;
      float mx = -INFINITY;
      for (int nn = half * 256; nn < half * 256 + 256; ++nn) mx = fmaxf(mx, rp[nn]);
      mx = fmaxf(mx, __shfl_xor(mx, 16, 32));
      float sum = 0.f;
      for (int nn = half * 256; nn < half * 256 + 256; ++nn) {
        float e = __expf(rp[nn] - mx);
        sum += e;
        rp[nn] = e;
      }
      sum += __shfl_xor(sum, 16, 32);
      float inv = 1.0f / sum;
      _Float16* pp = mp + row * Nn;
      for (int nn = half * 256; nn < half * 256 + 256; ++nn)
        pp[nn] = (_Float16)(rp[nn] * inv);
    }
    __syncthreads();
    // out tile = P @ V  (K = N = 512)
    v8f co = {};
    for (int kb = 0; kb < Nn; kb += 32) {
      v16h a = ld_a16(mp, Nn, kb, lane);
      v16h bv = ld_bt16(vshT, Nn, kb, lane);   // col = d, K = n (contiguous in vshT row)
      co = WMMA16(a, bv, co);
    }
    int m0 = (lane >> 4) * 8, col = lane & 15;
#pragma unroll
    for (int r = 0; r < 8; ++r)
      out16[((size_t)(b * Gg + g0 + m0 + r)) * EMB + h * DK + col] = (_Float16)co[r];
    __syncthreads();
  }
}

// ---------------------------------------------------------------- kernel 6: top-2 gating
__global__ void gate_kernel(const _Float16* out16, const float* gate_w, const float* prefgate,
                            float* gates) {
  int t = blockIdx.x * blockDim.x + threadIdx.x;
  if (t >= Bb * Gg) return;
  float lg[Ee];
#pragma unroll
  for (int e = 0; e < Ee; ++e) lg[e] = prefgate[e];
  const _Float16* x = out16 + (size_t)t * EMB;
  for (int d = 0; d < EMB; ++d) {
    float xv = (float)x[d];
#pragma unroll
    for (int e = 0; e < Ee; ++e) lg[e] += xv * gate_w[d * Ee + e];
  }
  int i0 = 0;
#pragma unroll
  for (int e = 1; e < Ee; ++e) if (lg[e] > lg[i0]) i0 = e;
  int i1 = (i0 == 0) ? 1 : 0;
#pragma unroll
  for (int e = 0; e < Ee; ++e) if (e != i0 && lg[e] > lg[i1]) i1 = e;
  float e1 = __expf(lg[i1] - lg[i0]);   // top-1 exp == 1
  float inv = 1.0f / (1.0f + e1);
#pragma unroll
  for (int e = 0; e < Ee; ++e)
    gates[(size_t)t * Ee + e] = (e == i0) ? inv : ((e == i1) ? e1 * inv : 0.f);
}

// ---------------------------------------------------------------- kernel 7: MoE (dense, gate-weighted)
// one block per 64-token tile; 8 waves; LDS: x(16K) + hdn(64K) + acc(32K) + gates(2K) = 114KB
__global__ void moe_kernel(const _Float16* out16, const float* gates, const _Float16* w1,
                           const float* b1, const _Float16* w2, const float* b2,
                           _Float16* moe16) {
  extern __shared__ char smem[];
  _Float16* xt  = (_Float16*)smem;                   // [64][128]
  _Float16* hdn = xt + 64 * EMB;                     // [64][512]
  float*    acc = (float*)(hdn + 64 * FH);           // [64][128]
  float*    gsh = acc + 64 * EMB;                    // [64][8]
  int t0 = blockIdx.x * 64;
  int tid = threadIdx.x, wave = tid >> 5, lane = tid & 31;
  // activation tile: contiguous -> async copy to LDS
  {
    unsigned xt_off = (unsigned)(uintptr_t)xt;
    const _Float16* Xg = out16 + (size_t)t0 * EMB;
    for (int c = tid; c < (64 * EMB * 2) / 16; c += blockDim.x)  // 1024 chunks
      async_ld_b128(xt_off + (unsigned)c * 16, Xg + (size_t)c * 8);
  }
  for (int i = tid; i < 64 * EMB; i += blockDim.x) acc[i] = 0.f;
  for (int i = tid; i < 64 * Ee; i += blockDim.x) gsh[i] = gates[(size_t)t0 * Ee + i];
  wait_async0();
  __syncthreads();

  for (int e = 0; e < Ee; ++e) {
    const _Float16* W1 = w1 + (size_t)e * FH * EMB;
    const float*    B1 = b1 + e * FH;
    const _Float16* W2 = w2 + (size_t)e * EMB * FH;
    const float*    B2 = b2 + e * EMB;
    // pull this expert's W2 toward the caches while the W1 GEMM runs
    for (int p = tid; p < (EMB * FH * 2) / 256; p += blockDim.x)
      __builtin_prefetch((const char*)W2 + (size_t)p * 256, 0, 2);
    // hdn = relu(x @ W1^T + b1): 4 m-tiles x 32 n-tiles
    for (int tt = wave; tt < 128; tt += 8) {
      int mt = tt >> 5, nt = tt & 31;
      v8f c = {};
#pragma unroll
      for (int kb = 0; kb < EMB; kb += 32) {
        v16h a = ld_a16(xt + mt * 16 * EMB, EMB, kb, lane);
        v16h b = ld_bt16(W1 + (size_t)(nt * 16) * EMB, EMB, kb, lane);
        c = WMMA16(a, b, c);
      }
      int m0 = (lane >> 4) * 8, n = lane & 15;
#pragma unroll
      for (int r = 0; r < 8; ++r) {
        float v = c[r] + B1[nt * 16 + n];
        hdn[(mt * 16 + m0 + r) * FH + nt * 16 + n] = (_Float16)fmaxf(v, 0.f);
      }
    }
    __syncthreads();
    // y = hdn @ W2^T + b2 ; acc += gate * y : 4 m-tiles x 8 o-tiles
    for (int tt = wave; tt < 32; tt += 8) {
      int mt = tt >> 3, ot = tt & 7;
      v8f c = {};
      for (int kb = 0; kb < FH; kb += 32) {
        v16h a = ld_a16(hdn + mt * 16 * FH, FH, kb, lane);
        v16h b = ld_bt16(W2 + (size_t)(ot * 16) * FH, FH, kb, lane);
        c = WMMA16(a, b, c);
      }
      int m0 = (lane >> 4) * 8, n = lane & 15;
#pragma unroll
      for (int r = 0; r < 8; ++r) {
        int m = mt * 16 + m0 + r;
        float y = c[r] + B2[ot * 16 + n];
        acc[m * EMB + ot * 16 + n] += gsh[m * Ee + e] * y;
      }
    }
    __syncthreads();
  }
  for (int i = tid; i < 64 * EMB; i += blockDim.x)
    moe16[(size_t)t0 * EMB + i] = (_Float16)acc[i];
}

// ---------------------------------------------------------------- kernel 8: combine + clipped masked softmax
// one block per (b, 16-row g-tile); 4 waves
__global__ void final_kernel(const _Float16* moe16, const _Float16* Wc16, const _Float16* enc16,
                             const float* mask, float* probs) {
  __shared__ _Float16 mh[16 * EMB];     // 4 KB
  __shared__ float    sb[16 * Nn];      // 32 KB
  int b = blockIdx.x >> 5, gt = blockIdx.x & 31, g0 = gt * 16;
  int tid = threadIdx.x, wave = tid >> 5, lane = tid & 31;
  const _Float16* X = moe16 + ((size_t)(b * Gg + g0)) * EMB;
  // mh = moe @ Wc^T : 8 o-tiles
  for (int ot = wave; ot < 8; ot += 4) {
    v8f c = {};
#pragma unroll
    for (int kb = 0; kb < EMB; kb += 32) {
      v16h a  = ld_a16(X, EMB, kb, lane);
      v16h bw = ld_bt16(Wc16 + (size_t)(ot * 16) * EMB, EMB, kb, lane);
      c = WMMA16(a, bw, c);
    }
    int m0 = (lane >> 4) * 8, n = lane & 15;
#pragma unroll
    for (int r = 0; r < 8; ++r) mh[(m0 + r) * EMB + ot * 16 + n] = (_Float16)c[r];
  }
  __syncthreads();
  const _Float16* E  = enc16 + (size_t)b * Nn * EMB;
  const float*    Mk = mask + ((size_t)(b * Gg + g0)) * Nn;
  // sc = mh . enc^T : 32 n-tiles
  for (int nt = wave; nt < 32; nt += 4) {
    v8f c = {};
#pragma unroll
    for (int kb = 0; kb < EMB; kb += 32) {
      v16h a  = ld_a16(mh, EMB, kb, lane);
      v16h be = ld_bt16(E + (size_t)(nt * 16) * EMB, EMB, kb, lane);
      c = WMMA16(a, be, c);
    }
    int m0 = (lane >> 4) * 8, n = lane & 15;
#pragma unroll
    for (int r = 0; r < 8; ++r) {
      int m = m0 + r;
      float sc = c[r] * INV_SQRT_EMB;
      sb[m * Nn + nt * 16 + n] = CLIP * tanhf(sc) + Mk[(size_t)m * Nn + nt * 16 + n];
    }
  }
  __syncthreads();
  if (tid < 16) {
    float* rp = sb + tid * Nn;
    float mx = -INFINITY;
    for (int n = 0; n < Nn; ++n) mx = fmaxf(mx, rp[n]);
    float s = 0.f;
    for (int n = 0; n < Nn; ++n) { float e = __expf(rp[n] - mx); s += e; rp[n] = e; }
    float inv = 1.0f / s;
    float* outp = probs + ((size_t)(b * Gg + g0 + tid)) * Nn;
    for (int n = 0; n < Nn; ++n) outp[n] = rp[n] * inv;
  }
}

// ---------------------------------------------------------------- launch
extern "C" void kernel_launch(void* const* d_in, const int* in_sizes, int n_in,
                              void* d_out, int out_size, void* d_ws, size_t ws_size,
                              hipStream_t stream) {
  const float* pref        = (const float*)d_in[0];
  const float* graph       = (const float*)d_in[1];
  const float* capacity    = (const float*)d_in[2];
  const float* enc         = (const float*)d_in[3];
  const float* ninf_mask   = (const float*)d_in[4];
  const float* fc1_w       = (const float*)d_in[5];
  const float* fc1_b       = (const float*)d_in[6];
  const float* fc2_w       = (const float*)d_in[7];
  const float* fc2_b       = (const float*)d_in[8];
  const float* fc3_w       = (const float*)d_in[9];
  const float* fc3_b       = (const float*)d_in[10];
  const float* hWq         = (const float*)d_in[11];
  const float* hWk         = (const float*)d_in[12];
  const float* hWv         = (const float*)d_in[13];
  const float* hWc         = (const float*)d_in[14];
  const float* gate_w      = (const float*)d_in[15];
  const float* gate_pref_w = (const float*)d_in[16];
  const float* e_w1        = (const float*)d_in[17];
  const float* e_b1        = (const float*)d_in[18];
  const float* e_w2        = (const float*)d_in[19];
  const float* e_b2        = (const float*)d_in[20];
  float* probs = (float*)d_out;

  char* ws = (char*)d_ws;
  size_t off = 0;
  auto carve = [&](size_t bytes) -> char* {
    char* p = ws + off;
    off = (off + bytes + 255) & ~(size_t)255;
    return p;
  };
  float*     mid      = (float*)carve(8 * 4);
  float*     prefgate = (float*)carve(8 * 4);
  float*     Wq       = (float*)carve((size_t)(EMB + 1) * EMB * 4);
  float*     qbase    = (float*)carve((size_t)Bb * EMB * 4);
  _Float16*  Wk16     = (_Float16*)carve((size_t)EMB * EMB * 2);
  _Float16*  Wv16     = (_Float16*)carve((size_t)EMB * EMB * 2);
  _Float16*  Wc16     = (_Float16*)carve((size_t)EMB * EMB * 2);
  _Float16*  enc16    = (_Float16*)carve(ENC_CNT * 2);
  _Float16*  w1_16    = (_Float16*)carve(W1_CNT * 2);
  _Float16*  w2_16    = (_Float16*)carve(W2_CNT * 2);
  _Float16*  q16      = (_Float16*)carve(Q_CNT * 2);
  _Float16*  K16      = (_Float16*)carve((size_t)Bb * Nn * EMB * 2);
  _Float16*  V16      = (_Float16*)carve((size_t)Bb * Nn * EMB * 2);
  _Float16*  out16    = (_Float16*)carve((size_t)Bb * Gg * EMB * 2);
  float*     gates    = (float*)carve((size_t)Bb * Gg * Ee * 4);
  _Float16*  moe16    = (_Float16*)carve((size_t)Bb * Gg * EMB * 2);

  hyper_kernel<<<1, 256, 0, stream>>>(pref, fc1_w, fc1_b, fc2_w, fc2_b, fc3_w, fc3_b,
                                      hWq, hWk, hWv, hWc, gate_pref_w,
                                      mid, prefgate, Wq, Wk16, Wv16, Wc16);
  qbase_kernel<<<Bb, EMB, 0, stream>>>(Wq, graph, qbase);
  convert_kernel<<<4096, 256, 0, stream>>>(enc, e_w1, e_w2, qbase, capacity, Wq,
                                           enc16, w1_16, w2_16, q16);
  kv_kernel<<<2048, 256, 0, stream>>>(enc16, Wk16, Wv16, K16, V16);

  constexpr int ATTN_LDS = Nn * DK * 2 + DK * Nn * 2 + 4 * 16 * Nn * 4 + 4 * 16 * Nn * 2; // 229376
  attn_kernel<<<Bb * Hh, 128, ATTN_LDS, stream>>>(q16, K16, V16, ninf_mask, out16);

  gate_kernel<<<(Bb * Gg + 255) / 256, 256, 0, stream>>>(out16, gate_w, prefgate, gates);

  constexpr int MOE_LDS = 64 * EMB * 2 + 64 * FH * 2 + 64 * EMB * 4 + 64 * Ee * 4; // 116736
  moe_kernel<<<(Bb * Gg) / 64, 256, MOE_LDS, stream>>>(out16, gates, w1_16, e_b1,
                                                       w2_16, e_b2, moe16);

  final_kernel<<<Bb * 32, 128, 0, stream>>>(moe16, Wc16, enc16, ninf_mask, probs);

  (void)in_sizes; (void)n_in; (void)out_size; (void)ws_size;
}
